// SetAbstraction_6158983102671
// MI455X (gfx1250) — matrix operand order repaired and checked by reading, not compiled, not run on hardware
//
#include <hip/hip_runtime.h>
#include <hip/hip_bf16.h>

// ---------------- types for WMMA fragments (wave32, CDNA5) ----------------
typedef __attribute__((ext_vector_type(16))) __bf16 v16bf;
typedef __attribute__((ext_vector_type(8)))  __bf16 v8bf;
typedef __attribute__((ext_vector_type(8)))  float  v8f;

// ---------------- problem constants ----------------
#define BATCH   16
#define NPTS    4096
#define SPTS    1024
#define KSAMP   32
#define DFEAT   64
#define MTOT    (BATCH * SPTS * KSAMP)   // 524288 GEMM rows
#define KP0     96                       // 67 padded to 3*32
#define KP12    64                       // 2*32
#define R2      0.04f
#define BN_EPS  1e-5f

// workspace layout (bytes)
#define XB_BYTES   ((size_t)MTOT * KP0 * 2)            // 100,663,296 (bf16 activations)
#define YA_BYTES   ((size_t)MTOT * 128 * 4)            // 268,435,456 (f32 GEMM out)
#define IDX_BYTES  ((size_t)MTOT * 4)                  // 2,097,152
#define OFF_YA     (XB_BYTES)
#define OFF_IDX    (XB_BYTES + YA_BYTES)
#define OFF_STATS  (OFF_IDX + IDX_BYTES)
#define STATS_F    (3 * 512)                           // per layer: sum[128],sq[128],scale[128],shift[128]
#define OFF_WB0    (OFF_STATS + STATS_F * 4)
#define OFF_WB1    (OFF_WB0 + (size_t)64  * KP0  * 2)
#define OFF_WB2    (OFF_WB1 + (size_t)64  * KP12 * 2)

// ---------------- zero the BN accumulators each call ----------------
__global__ void k_zero_stats(float* stats) {
    float* base = stats + (size_t)blockIdx.x * 512;
    base[threadIdx.x] = 0.f;           // sum
    base[128 + threadIdx.x] = 0.f;     // sumsq
}

// ---------------- convert fp32 weights -> padded row-major bf16 ----------------
__global__ void k_convert_w(const float* __restrict__ W, __bf16* __restrict__ Wb,
                            int O, int Cin, int Kpad) {
    int t = blockIdx.x * blockDim.x + threadIdx.x;
    if (t >= O * Kpad) return;
    int o = t / Kpad, c = t - o * Kpad;
    Wb[t] = (c < Cin) ? (__bf16)W[o * Cin + c] : (__bf16)0.f;
}

// ---------------- farthest point sampling: 1 block per batch ----------------
// xyz tile lives in LDS (48 KB), running min-distance in 4 regs/thread.
__global__ void __launch_bounds__(1024) k_fps(const float* __restrict__ xyz,
                                              float* __restrict__ out_xyz) {
    __shared__ float sx[NPTS], sy[NPTS], sz[NPTS];
    __shared__ float rv[32];
    __shared__ int   ri[32];
    __shared__ int   s_far;
    const int b = blockIdx.x, tid = threadIdx.x;
    const int lane = tid & 31, wid = tid >> 5;
    const float* px = xyz + (size_t)b * NPTS * 3;
    for (int i = tid; i < NPTS; i += 1024) {
        sx[i] = px[i * 3 + 0]; sy[i] = px[i * 3 + 1]; sz[i] = px[i * 3 + 2];
    }
    float dloc[4];
#pragma unroll
    for (int j = 0; j < 4; ++j) dloc[j] = 1e10f;
    __syncthreads();

    int far = 0;
    for (int it = 0; it < SPTS; ++it) {
        const float cx = sx[far], cy = sy[far], cz = sz[far];
        if (tid == 0) {
            float* o = out_xyz + ((size_t)b * SPTS + it) * 3;
            o[0] = cx; o[1] = cy; o[2] = cz;
        }
        float bv = -1.f; int bi = 0;
#pragma unroll
        for (int j = 0; j < 4; ++j) {
            const int i = tid + j * 1024;
            const float dx = sx[i] - cx, dy = sy[i] - cy, dz = sz[i] - cz;
            const float d  = dx * dx + dy * dy + dz * dz;
            const float nd = fminf(dloc[j], d);
            dloc[j] = nd;
            if (nd > bv || (nd == bv && i < bi)) { bv = nd; bi = i; }
        }
        // wave32 argmax (prefer smaller index on ties, matching jnp.argmax)
        for (int off = 16; off > 0; off >>= 1) {
            const float ov = __shfl_xor(bv, off, 32);
            const int   oi = __shfl_xor(bi, off, 32);
            if (ov > bv || (ov == bv && oi < bi)) { bv = ov; bi = oi; }
        }
        if (lane == 0) { rv[wid] = bv; ri[wid] = bi; }
        __syncthreads();
        if (wid == 0) {
            float v = rv[lane]; int ix = ri[lane];
            for (int off = 16; off > 0; off >>= 1) {
                const float ov = __shfl_xor(v, off, 32);
                const int   oi = __shfl_xor(ix, off, 32);
                if (ov > v || (ov == v && oi < ix)) { v = ov; ix = oi; }
            }
            if (lane == 0) s_far = ix;
        }
        __syncthreads();
        far = s_far;
    }
}

// ---------------- ball query: one wave per (b,s); dists in LDS ----------------
#define BQ_WAVES 2
__global__ void __launch_bounds__(32 * BQ_WAVES) k_ballquery(const float* __restrict__ xyz,
                                                             const float* __restrict__ new_xyz,
                                                             int* __restrict__ idxbuf) {
    __shared__ float sd[BQ_WAVES][NPTS];
    const int w = threadIdx.x >> 5, lane = threadIdx.x & 31;
    const int p = blockIdx.x * BQ_WAVES + w;        // (b,s) flat
    const int b = p >> 10;
    const float* q = new_xyz + (size_t)p * 3;
    const float qx = q[0], qy = q[1], qz = q[2];
    const float* px = xyz + (size_t)b * NPTS * 3;

    float lval = 3.4e38f; int lidx = lane;
    for (int j = 0; j < 128; ++j) {
        const int n = j * 32 + lane;
        const float dx = px[n * 3 + 0] - qx;
        const float dy = px[n * 3 + 1] - qy;
        const float dz = px[n * 3 + 2] - qz;
        const float d  = dx * dx + dy * dy + dz * dz;
        sd[w][n] = d;
        if (d < lval) { lval = d; lidx = n; }
    }
    int near = 0;
    int* ob = idxbuf + (size_t)p * KSAMP;
    for (int r = 0; r < KSAMP; ++r) {
        float mv = lval; int mi = lidx;
        for (int off = 16; off > 0; off >>= 1) {
            const float ov = __shfl_xor(mv, off, 32);
            const int   oi = __shfl_xor(mi, off, 32);
            if (ov < mv || (ov == mv && oi < mi)) { mv = ov; mi = oi; }
        }
        if (r == 0) near = mi;
        if (lane == r) ob[r] = (mv > R2) ? near : mi;   // out-of-ball -> nearest
        if (lane == (mi & 31)) {                         // winner lane removes + rescans
            sd[w][mi] = 3.4e38f;
            lval = 3.4e38f; lidx = lane;
            for (int j = 0; j < 128; ++j) {
                const int n = j * 32 + lane;
                const float v = sd[w][n];
                if (v < lval) { lval = v; lidx = n; }
            }
        }
    }
}

// ---------------- gather + concat -> bf16 x0 [M, 96] ----------------
__global__ void k_gather(const float* __restrict__ xyz, const float* __restrict__ pts,
                         const float* __restrict__ new_xyz, const int* __restrict__ idxbuf,
                         __bf16* __restrict__ xb) {
    const int row = blockIdx.x * blockDim.x + threadIdx.x;   // < MTOT
    const int b = row >> 15;
    const int s = (row >> 5) & 1023;
    const int n = idxbuf[row];
    const float* q = new_xyz + ((size_t)(b << 10) + s) * 3;
    const float* g = xyz + ((size_t)(b << 12) + n) * 3;
    __bf16* o = xb + (size_t)row * KP0;
    o[0] = (__bf16)(g[0] - q[0]);
    o[1] = (__bf16)(g[1] - q[1]);
    o[2] = (__bf16)(g[2] - q[2]);
    const float* pp = pts + ((size_t)(b << 12) + n) * DFEAT;
#pragma unroll 8
    for (int c = 0; c < DFEAT; ++c) o[3 + c] = (__bf16)pp[c];
#pragma unroll
    for (int c = 3 + DFEAT; c < KP0; ++c) o[c] = (__bf16)0.f;
}

// ---------------- WMMA GEMM + bias + BN partial stats ----------------
// Y[M,Nout] = A[M,Kpad] * Wb[Nout,Kpad]^T + bias ; accumulate per-channel sum/sumsq.
// 256 threads = 8 waves; each wave owns 16 rows. A fragments for all K tiles are
// loaded ONCE into registers, then reused across every N tile (removes the NT-fold
// A reload of the previous version).
template <int KT, int NT>
__global__ void __launch_bounds__(256) k_gemm(const __bf16* __restrict__ A,
                                              const __bf16* __restrict__ Wb,
                                              const float* __restrict__ bias,
                                              float* __restrict__ Y,
                                              float* __restrict__ gsum,
                                              float* __restrict__ gsq) {
    constexpr int Kpad = KT * 32;
    constexpr int Nout = NT * 16;
    __shared__ float s_sum[128], s_sq[128];
    const int tid = threadIdx.x;
    if (tid < 128) { s_sum[tid] = 0.f; s_sq[tid] = 0.f; }
    __syncthreads();

    const int wid  = tid >> 5, lane = tid & 31;
    const int lh   = lane >> 4;          // half-wave select
    const int l15  = lane & 15;
    const size_t m_base = ((size_t)blockIdx.x * 8 + wid) * 16;
    const __bf16* pa_row = A + (m_base + l15) * Kpad + lh * 8;

    // pull the (tiny, L2-resident) weight tile toward the WGP
    __builtin_prefetch(Wb + (size_t)l15 * Kpad, 0, 3);

    // load all A fragments once (KT * 8 VGPRs)
    v16bf av[KT];
#pragma unroll
    for (int kt = 0; kt < KT; ++kt) {
        const __bf16* pa = pa_row + kt * 32;
        const v8bf alo = *(const v8bf*)(pa);
        const v8bf ahi = *(const v8bf*)(pa + 16);
        av[kt] = __builtin_shufflevector(alo, ahi,
            0, 1, 2, 3, 4, 5, 6, 7, 8, 9, 10, 11, 12, 13, 14, 15);
    }

#pragma unroll
    for (int nt = 0; nt < NT; ++nt) {
        v8f acc = {0.f, 0.f, 0.f, 0.f, 0.f, 0.f, 0.f, 0.f};
        const int col = nt * 16 + l15;
        const __bf16* pb_row = Wb + (size_t)col * Kpad + lh * 8;
#pragma unroll
        for (int kt = 0; kt < KT; ++kt) {
            const __bf16* pb = pb_row + kt * 32;
            const v8bf blo = *(const v8bf*)(pb);
            const v8bf bhi = *(const v8bf*)(pb + 16);
            const v16bf bv = __builtin_shufflevector(blo, bhi,
                0, 1, 2, 3, 4, 5, 6, 7, 8, 9, 10, 11, 12, 13, 14, 15);
            acc = __builtin_amdgcn_wmma_f32_16x16x32_bf16(
                false, av[kt], false, bv, (short)0, acc, false, false);
        }
        const float bb = bias[col];
        float ls = 0.f, lq = 0.f;
#pragma unroll
        for (int i = 0; i < 8; ++i) {
            const float v = acc[i] + bb;
            const size_t r = m_base + i + lh * 8;
            Y[r * Nout + col] = v;
            ls += v; lq += v * v;
        }
        atomicAdd(&s_sum[col], ls);
        atomicAdd(&s_sq[col], lq);
    }
    __syncthreads();
    if (tid < Nout) {
        atomicAdd(&gsum[tid], s_sum[tid]);
        atomicAdd(&gsq[tid],  s_sq[tid]);
    }
}

// ---------------- BN stats -> scale/shift ----------------
__global__ void k_finalize(const float* __restrict__ gsum, const float* __restrict__ gsq,
                           const float* __restrict__ gamma, const float* __restrict__ beta,
                           float* __restrict__ scale, float* __restrict__ shift, int O) {
    const int t = threadIdx.x;
    if (t >= O) return;
    const float inv = 1.f / (float)MTOT;
    const float mu  = gsum[t] * inv;
    const float var = gsq[t] * inv - mu * mu;
    const float sc  = gamma[t] * rsqrtf(var + BN_EPS);
    scale[t] = sc;
    shift[t] = beta[t] - mu * sc;
}

// ---------------- fused BN + ReLU + cast to bf16 (next layer input) ----------------
__global__ void k_bnrelu(const float* __restrict__ Y, const float* __restrict__ scale,
                         const float* __restrict__ shift, __bf16* __restrict__ xb) {
    const size_t t = (size_t)blockIdx.x * blockDim.x + threadIdx.x;  // < MTOT*64
    const int c = (int)(t & 63);
    float v = scale[c] * Y[t] + shift[c];
    v = fmaxf(v, 0.f);
    xb[t] = (__bf16)v;
}

// ---------------- BN3 + ReLU + max-pool over K ----------------
__global__ void k_maxpool(const float* __restrict__ Y, const float* __restrict__ scale,
                          const float* __restrict__ shift, float* __restrict__ outp) {
    const int t  = blockIdx.x * blockDim.x + threadIdx.x;  // < B*S*128
    const int c  = t & 127;
    const int bs = t >> 7;
    const float* p = Y + (size_t)bs * KSAMP * 128 + c;
    const float sc = scale[c], sh = shift[c];
    float m = 0.f;  // ReLU output is >= 0, so 0 is a safe identity
#pragma unroll
    for (int k = 0; k < KSAMP; ++k) {
        const float v = fmaxf(sc * p[k * 128] + sh, 0.f);
        m = fmaxf(m, v);
    }
    outp[t] = m;
}

// ---------------- host-side orchestration ----------------
extern "C" void kernel_launch(void* const* d_in, const int* in_sizes, int n_in,
                              void* d_out, int out_size, void* d_ws, size_t ws_size,
                              hipStream_t stream) {
    (void)in_sizes; (void)n_in; (void)out_size; (void)ws_size;
    const float* xyz = (const float*)d_in[0];
    const float* pts = (const float*)d_in[1];
    const float* W0  = (const float*)d_in[2];
    const float* b0  = (const float*)d_in[3];
    const float* g0  = (const float*)d_in[4];
    const float* bt0 = (const float*)d_in[5];
    const float* W1  = (const float*)d_in[6];
    const float* b1  = (const float*)d_in[7];
    const float* g1  = (const float*)d_in[8];
    const float* bt1 = (const float*)d_in[9];
    const float* W2  = (const float*)d_in[10];
    const float* b2  = (const float*)d_in[11];
    const float* g2  = (const float*)d_in[12];
    const float* bt2 = (const float*)d_in[13];

    float* out        = (float*)d_out;
    float* new_xyz    = out;                              // [B,S,3]
    float* new_points = out + (size_t)BATCH * SPTS * 3;   // [B,S,128]

    char*   ws    = (char*)d_ws;
    __bf16* xb    = (__bf16*)ws;
    float*  yA    = (float*)(ws + OFF_YA);
    int*    idxb  = (int*)(ws + OFF_IDX);
    float*  stats = (float*)(ws + OFF_STATS);
    __bf16* wb0   = (__bf16*)(ws + OFF_WB0);
    __bf16* wb1   = (__bf16*)(ws + OFF_WB1);
    __bf16* wb2   = (__bf16*)(ws + OFF_WB2);
    float* st0 = stats, *st1 = stats + 512, *st2 = stats + 1024;

    k_zero_stats<<<3, 128, 0, stream>>>(stats);
    k_convert_w<<<(64 * KP0  + 255) / 256, 256, 0, stream>>>(W0, wb0, 64,  67, KP0);
    k_convert_w<<<(64 * KP12 + 255) / 256, 256, 0, stream>>>(W1, wb1, 64,  64, KP12);
    k_convert_w<<<(128 * KP12 + 255) / 256, 256, 0, stream>>>(W2, wb2, 128, 64, KP12);

    k_fps<<<BATCH, 1024, 0, stream>>>(xyz, new_xyz);
    k_ballquery<<<(BATCH * SPTS) / BQ_WAVES, 32 * BQ_WAVES, 0, stream>>>(xyz, new_xyz, idxb);
    k_gather<<<MTOT / 256, 256, 0, stream>>>(xyz, pts, new_xyz, idxb, xb);

    // layer 1: [M,96] x [64,96]^T
    k_gemm<3, 4><<<MTOT / 128, 256, 0, stream>>>(xb, wb0, b0, yA, st0, st0 + 128);
    k_finalize<<<1, 128, 0, stream>>>(st0, st0 + 128, g0, bt0, st0 + 256, st0 + 384, 64);
    k_bnrelu<<<(int)(((size_t)MTOT * 64) / 256), 256, 0, stream>>>(yA, st0 + 256, st0 + 384, xb);

    // layer 2: [M,64] x [64,64]^T
    k_gemm<2, 4><<<MTOT / 128, 256, 0, stream>>>(xb, wb1, b1, yA, st1, st1 + 128);
    k_finalize<<<1, 128, 0, stream>>>(st1, st1 + 128, g1, bt1, st1 + 256, st1 + 384, 64);
    k_bnrelu<<<(int)(((size_t)MTOT * 64) / 256), 256, 0, stream>>>(yA, st1 + 256, st1 + 384, xb);

    // layer 3: [M,64] x [128,64]^T
    k_gemm<2, 8><<<MTOT / 128, 256, 0, stream>>>(xb, wb2, b2, yA, st2, st2 + 128);
    k_finalize<<<1, 128, 0, stream>>>(st2, st2 + 128, g2, bt2, st2 + 256, st2 + 384, 128);

    k_maxpool<<<(BATCH * SPTS * 128) / 256, 256, 0, stream>>>(yA, st2 + 256, st2 + 384, new_points);
}
